// MultiHeadAttention_9964324127450
// MI455X (gfx1250) — matrix operand order repaired and checked by reading, not compiled, not run on hardware
//
#include <hip/hip_runtime.h>
#include <hip/hip_bf16.h>

// ---------------------------------------------------------------------------
// MHA forward for MI455X (gfx1250, wave32, WMMA f32_16x16x32_f16)
// B=2, S=2048, D_MODEL=1024, H=16, D=64
// ---------------------------------------------------------------------------

typedef _Float16 half8  __attribute__((ext_vector_type(8)));
typedef _Float16 half16 __attribute__((ext_vector_type(16)));
typedef float    float8 __attribute__((ext_vector_type(8)));

#define BQ 2
#define SQ 2048
#define DM 1024
#define HQ 16
#define DH 64
#define ROWS (BQ*SQ)          // 4096

__device__ __forceinline__ half16 combine(half8 lo, half8 hi) {
    half16 r;
#pragma unroll
    for (int i = 0; i < 8; ++i) { r[i] = lo[i]; r[i + 8] = hi[i]; }
    return r;
}

__device__ __forceinline__ half8 ld8(const _Float16* p) {
    return *(const half8*)p;
}

// reduce across the 16 lanes that hold one matrix row in C-layout
__device__ __forceinline__ float redmax16(float v) {
#pragma unroll
    for (int m = 1; m < 16; m <<= 1) v = fmaxf(v, __shfl_xor(v, m, 32));
    return v;
}
__device__ __forceinline__ float redsum16(float v) {
#pragma unroll
    for (int m = 1; m < 16; m <<= 1) v += __shfl_xor(v, m, 32);
    return v;
}

// ---------------------------------------------------------------------------
// Phase 0: conversions / transposes into f16 workspace
// ---------------------------------------------------------------------------
__global__ void cvt_f32_f16(const float* __restrict__ in, _Float16* __restrict__ out, int n) {
    int i = blockIdx.x * blockDim.x + threadIdx.x;
    if (i < n) out[i] = (_Float16)in[i];
}

// [H,1024,64] -> [H,64,1024]
__global__ void cvt_whT(const float* __restrict__ in, _Float16* __restrict__ out) {
    int i = blockIdx.x * blockDim.x + threadIdx.x;   // < 16*1024*64
    int h = i >> 16, r = i & 65535, m = r >> 6, d = r & 63;
    out[(h << 16) + (d << 10) + m] = (_Float16)in[i];
}

// [1024,1024] -> transpose
__global__ void cvt_woT(const float* __restrict__ in, _Float16* __restrict__ out) {
    int i = blockIdx.x * blockDim.x + threadIdx.x;   // < 1024*1024
    int m = i >> 10, n = i & 1023;
    out[(n << 10) + m] = (_Float16)in[i];
}

// ---------------------------------------------------------------------------
// Phase 1: QKV projections.  One wave per (proj p, head h, 16-row tile rt).
// Computes a 16x64 tile:  C = X[rt*16:+16, :] @ W_p,h  (K = 1024)
// Q,K stored [B,H,S,64]; V stored transposed VT [B,H,64,S].
// ---------------------------------------------------------------------------
__global__ __launch_bounds__(32) void qkv_gemm(
    const _Float16* __restrict__ xh,      // [4096,1024]
    const _Float16* __restrict__ wT,      // [3][16][64][1024]
    const float* __restrict__ bq, const float* __restrict__ bk, const float* __restrict__ bv,
    _Float16* __restrict__ Q, _Float16* __restrict__ K, _Float16* __restrict__ VT)
{
    const int lane = threadIdx.x & 31;
    const int hf   = lane >> 4;     // k-half selector
    const int lr   = lane & 15;

    const int rt = blockIdx.x;      // 0..255 row tile over B*S
    const int h  = blockIdx.y;      // head
    const int p  = blockIdx.z;      // 0=q 1=k 2=v

    const _Float16* W    = wT + p * (HQ * DH * DM) + h * (DH * DM);  // [64][1024]
    const float*    bias = (p == 0 ? bq : (p == 1 ? bk : bv)) + h * DH;

    float8 acc[4];
#pragma unroll
    for (int nt = 0; nt < 4; ++nt) acc[nt] = (float8){};

    const _Float16* Arow = xh + (rt * 16 + lr) * DM;

    for (int k0 = 0; k0 < DM; k0 += 32) {
        half16 a = combine(ld8(Arow + k0 + 16 * hf), ld8(Arow + k0 + 16 * hf + 8));
#pragma unroll
        for (int nt = 0; nt < 4; ++nt) {
            const _Float16* Bc = W + (nt * 16 + lr) * DM + k0 + 16 * hf;
            half16 b = combine(ld8(Bc), ld8(Bc + 8));
            acc[nt] = __builtin_amdgcn_wmma_f32_16x16x32_f16(
                false, a, false, b, (short)0, acc[nt], false, false);
        }
    }

    const int bb = (rt * 16) >> 11;          // batch
    const int s0 = (rt * 16) & (SQ - 1);     // seq base within batch
    const int bh = bb * HQ + h;

#pragma unroll
    for (int nt = 0; nt < 4; ++nt) {
        const int n = nt * 16 + lr;
        const float bi = bias[n];
        if (p == 2) {
            half8 v;
#pragma unroll
            for (int i = 0; i < 8; ++i) v[i] = (_Float16)(acc[nt][i] + bi);
            // lane holds 8 consecutive s for fixed d = n
            *(half8*)(VT + ((size_t)bh * DH + n) * SQ + s0 + 8 * hf) = v;
        } else {
            _Float16* dst = (p == 0) ? Q : K;
#pragma unroll
            for (int i = 0; i < 8; ++i) {
                int s = s0 + i + 8 * hf;
                dst[((size_t)bh * SQ + s) * DH + n] = (_Float16)(acc[nt][i] + bi);
            }
        }
    }
}

// ---------------------------------------------------------------------------
// Phase 2: flash attention.  One wave per (bh, 16-row query tile).
// Online softmax over 64 column tiles of 32.  Output to U [B,S,1024] f16.
// ---------------------------------------------------------------------------
#define PPITCH 40   // halfs per row of the LDS P-staging tile (80B, 16B aligned)

__global__ __launch_bounds__(128) void attention(
    const _Float16* __restrict__ Q,   // [B,H,S,64]
    const _Float16* __restrict__ K,   // [B,H,S,64]
    const _Float16* __restrict__ VT,  // [B,H,64,S]
    _Float16* __restrict__ U)         // [B,S,1024]
{
    __shared__ _Float16 pbuf[4][16 * PPITCH];

    const int tid  = threadIdx.x;
    const int wid  = tid >> 5;
    const int lane = tid & 31;
    const int hf   = lane >> 4;
    const int lr   = lane & 15;

    const int rt = blockIdx.x * 4 + wid;   // 0..127 query row tile
    const int bh = blockIdx.y;             // 0..31
    const int bb = bh >> 4, h = bh & 15;

    const _Float16* Qp = Q  + (size_t)bh * SQ * DH;
    const _Float16* Kp = K  + (size_t)bh * SQ * DH;
    const _Float16* Vp = VT + (size_t)bh * DH * SQ;
    _Float16* pl = pbuf[wid];

    // Q A-fragments for this wave's 16 rows (d = 0..31 and 32..63)
    const _Float16* qrow = Qp + (rt * 16 + lr) * DH;
    const half16 aq0 = combine(ld8(qrow +      16 * hf), ld8(qrow +      16 * hf + 8));
    const half16 aq1 = combine(ld8(qrow + 32 + 16 * hf), ld8(qrow + 32 + 16 * hf + 8));

    float8 acc[4];
#pragma unroll
    for (int nt = 0; nt < 4; ++nt) acc[nt] = (float8){};
    float m_run[8], l_run[8];
#pragma unroll
    for (int i = 0; i < 8; ++i) { m_run[i] = -1e30f; l_run[i] = 0.0f; }

    for (int t0 = 0; t0 < SQ; t0 += 32) {
        // ---- scores: two 16x16 tiles over 32 key columns -------------------
        float8 s0 = (float8){}, s1 = (float8){};
        {
            const _Float16* kr = Kp + (t0 + lr) * DH;
            half16 bA = combine(ld8(kr +      16 * hf), ld8(kr +      16 * hf + 8));
            half16 bB = combine(ld8(kr + 32 + 16 * hf), ld8(kr + 32 + 16 * hf + 8));
            s0 = __builtin_amdgcn_wmma_f32_16x16x32_f16(false, aq0, false, bA, (short)0, s0, false, false);
            s0 = __builtin_amdgcn_wmma_f32_16x16x32_f16(false, aq1, false, bB, (short)0, s0, false, false);
        }
        {
            const _Float16* kr = Kp + (t0 + 16 + lr) * DH;
            half16 bA = combine(ld8(kr +      16 * hf), ld8(kr +      16 * hf + 8));
            half16 bB = combine(ld8(kr + 32 + 16 * hf), ld8(kr + 32 + 16 * hf + 8));
            s1 = __builtin_amdgcn_wmma_f32_16x16x32_f16(false, aq0, false, bA, (short)0, s1, false, false);
            s1 = __builtin_amdgcn_wmma_f32_16x16x32_f16(false, aq1, false, bB, (short)0, s1, false, false);
        }

        // ---- online softmax update ----------------------------------------
        float p0[8], p1[8];
#pragma unroll
        for (int i = 0; i < 8; ++i) {
            float a = s0[i] * 0.125f;          // 1/sqrt(64)
            float b = s1[i] * 0.125f;
            float rmax = redmax16(fmaxf(a, b));
            float mn   = fmaxf(m_run[i], rmax);
            float corr = __expf(m_run[i] - mn);
            m_run[i] = mn;
            p0[i] = __expf(a - mn);
            p1[i] = __expf(b - mn);
            l_run[i] = l_run[i] * corr + redsum16(p0[i] + p1[i]);
#pragma unroll
            for (int nt = 0; nt < 4; ++nt) acc[nt][i] *= corr;
        }

        // ---- C-layout -> A-layout transpose of P via LDS ------------------
#pragma unroll
        for (int i = 0; i < 8; ++i) {
            pl[(i + 8 * hf) * PPITCH + lr]      = (_Float16)p0[i];
            pl[(i + 8 * hf) * PPITCH + 16 + lr] = (_Float16)p1[i];
        }
        asm volatile("s_wait_dscnt 0" ::: "memory");
        half16 ap = combine(*(const half8*)(pl + lr * PPITCH + 8 * hf),
                            *(const half8*)(pl + lr * PPITCH + 16 + 8 * hf));

        // ---- acc += P @ V  (B-frags contiguous from V-transposed) ---------
#pragma unroll
        for (int nt = 0; nt < 4; ++nt) {
            const _Float16* vc = Vp + (nt * 16 + lr) * SQ + t0 + 8 * hf;
            half16 bv = combine(ld8(vc), ld8(vc + 16));
            acc[nt] = __builtin_amdgcn_wmma_f32_16x16x32_f16(
                false, ap, false, bv, (short)0, acc[nt], false, false);
        }
        asm volatile("s_wait_dscnt 0" ::: "memory");  // WAR: reads done before next-iter stores
    }

    // ---- normalize and store concat-head output -----------------------------
    float inv[8];
#pragma unroll
    for (int i = 0; i < 8; ++i) inv[i] = 1.0f / l_run[i];
#pragma unroll
    for (int nt = 0; nt < 4; ++nt) {
        const int col = h * DH + nt * 16 + lr;
#pragma unroll
        for (int i = 0; i < 8; ++i) {
            int s = rt * 16 + i + 8 * hf;
            U[((size_t)bb * SQ + s) * DM + col] = (_Float16)(acc[nt][i] * inv[i]);
        }
    }
}

// ---------------------------------------------------------------------------
// Phase 3: output projection  Y = U @ Wo + bo   (M=4096, N=1024, K=1024)
// One wave per 16x64 output tile.
// ---------------------------------------------------------------------------
__global__ __launch_bounds__(128) void out_proj(
    const _Float16* __restrict__ U,    // [4096,1024]
    const _Float16* __restrict__ WoT,  // [1024,1024] transposed
    const float* __restrict__ bo,
    float* __restrict__ out)           // [4096,1024]
{
    const int tid  = threadIdx.x;
    const int wid  = tid >> 5;
    const int lane = tid & 31;
    const int hf   = lane >> 4;
    const int lr   = lane & 15;

    const int mt = blockIdx.x;                 // 0..255
    const int ng = blockIdx.y * 4 + wid;       // 0..15, n_base = ng*64

    float8 acc[4];
#pragma unroll
    for (int nt = 0; nt < 4; ++nt) acc[nt] = (float8){};

    const _Float16* urow = U + (mt * 16 + lr) * DM;
    for (int k0 = 0; k0 < DM; k0 += 32) {
        half16 a = combine(ld8(urow + k0 + 16 * hf), ld8(urow + k0 + 16 * hf + 8));
#pragma unroll
        for (int nt = 0; nt < 4; ++nt) {
            const _Float16* wc = WoT + (ng * 64 + nt * 16 + lr) * DM + k0 + 16 * hf;
            half16 b = combine(ld8(wc), ld8(wc + 8));
            acc[nt] = __builtin_amdgcn_wmma_f32_16x16x32_f16(
                false, a, false, b, (short)0, acc[nt], false, false);
        }
    }

#pragma unroll
    for (int nt = 0; nt < 4; ++nt) {
        const int n = ng * 64 + nt * 16 + lr;
        const float bb = bo[n];
#pragma unroll
        for (int i = 0; i < 8; ++i) {
            out[(size_t)(mt * 16 + i + 8 * hf) * DM + n] = acc[nt][i] + bb;
        }
    }
}

// ---------------------------------------------------------------------------
// Launch
// ---------------------------------------------------------------------------
extern "C" void kernel_launch(void* const* d_in, const int* in_sizes, int n_in,
                              void* d_out, int out_size, void* d_ws, size_t ws_size,
                              hipStream_t stream) {
    (void)in_sizes; (void)n_in; (void)out_size; (void)ws_size;

    const float* x  = (const float*)d_in[0];
    const float* Wq = (const float*)d_in[1];
    const float* bq = (const float*)d_in[2];
    const float* Wk = (const float*)d_in[3];
    const float* bk = (const float*)d_in[4];
    const float* Wv = (const float*)d_in[5];
    const float* bv = (const float*)d_in[6];
    const float* Wo = (const float*)d_in[7];
    const float* bo = (const float*)d_in[8];
    float* out = (float*)d_out;

    // f16 workspace layout (element offsets)
    _Float16* ws = (_Float16*)d_ws;
    _Float16* xh   = ws;                       // 4096*1024        = 4194304
    _Float16* wT   = ws + 4194304;             // 3*16*64*1024     = 3145728
    _Float16* woT  = ws + 7340032;             // 1024*1024        = 1048576
    _Float16* Qb   = ws + 8388608;             // 2*16*2048*64     = 4194304
    _Float16* Kb   = ws + 12582912;            // 4194304
    _Float16* VTb  = ws + 16777216;            // 4194304
    _Float16* Ub   = ws + 20971520;            // 4096*1024        = 4194304

    const int T = 256;
    // Phase 0: conversions
    cvt_f32_f16<<<(ROWS * DM) / T, T, 0, stream>>>(x, xh, ROWS * DM);
    cvt_whT<<<(HQ * DM * DH) / T, T, 0, stream>>>(Wq, wT);
    cvt_whT<<<(HQ * DM * DH) / T, T, 0, stream>>>(Wk, wT + HQ * DM * DH);
    cvt_whT<<<(HQ * DM * DH) / T, T, 0, stream>>>(Wv, wT + 2 * HQ * DM * DH);
    cvt_woT<<<(DM * DM) / T, T, 0, stream>>>(Wo, woT);

    // Phase 1: QKV projections
    qkv_gemm<<<dim3(ROWS / 16, HQ, 3), 32, 0, stream>>>(xh, wT, bq, bk, bv, Qb, Kb, VTb);

    // Phase 2: attention (flash-style, online softmax)
    attention<<<dim3(SQ / 16 / 4, BQ * HQ), 128, 0, stream>>>(Qb, Kb, VTb, Ub);

    // Phase 3: output projection + bias
    out_proj<<<dim3(ROWS / 16, DM / 64 / 4), 128, 0, stream>>>(Ub, woT, bo, out);
}
